// MutualInformation_3264175145383
// MI455X (gfx1250) — compile-verified, matrix operand order checked by reading, not run on hardware
//
#include <hip/hip_runtime.h>
#include <hip/hip_bf16.h>

typedef __attribute__((ext_vector_type(16))) _Float16 v16h;
typedef __attribute__((ext_vector_type(8)))  float    v8f;

#define NPIX      65536          // 256*256 pixels per batch image
#define NBATCH    8
#define NBINS     256
#define NSLICE    4              // N-dim split for occupancy (deterministic partials)
#define SLICEPIX  (NPIX / NSLICE)
#define CHUNK     256            // pixels staged in LDS per pipeline stage
#define JSZ       ((size_t)NBATCH * NBINS * NBINS)   // elems per joint buffer
// exp(-0.5 d^2 / sigma^2) = exp2(C2 * d^2), sigma = 0.1  ->  C2 = -50 * log2(e)
#define C2        (-72.13475204444817f)
#define EPSF      1e-10f

// ---------------------------------------------------------------------------
// Kernel 1: partial joint GEMM via WMMA f16->f32, register-blocked 4x4 tiles,
// double-buffered async-to-LDS pixel staging (ASYNCcnt software pipeline).
// grid (2, 2, NBATCH*NSLICE), 128 threads = 4 waves in a 2x2 wave grid.
// Wave computes a 64x64 block: 4 A-frags x 4 B-frags -> 16 WMMAs per K=32.
// ---------------------------------------------------------------------------
__global__ __launch_bounds__(128)
void mi_joint_gemm(const float* __restrict__ x1, const float* __restrict__ x2,
                   float* __restrict__ jpart)
{
    __shared__ float lv1[2][CHUNK];
    __shared__ float lv2[2][CHUNK];

    const int lane  = threadIdx.x & 31;
    const int wave  = threadIdx.x >> 5;             // 0..3
    const int z     = blockIdx.z;
    const int b     = z >> 2;                       // batch
    const int slice = z & (NSLICE - 1);             // N-slice
    const int k0    = blockIdx.y * 128 + (wave >> 1) * 64;  // A-row (k-bin) base
    const int l0    = blockIdx.x * 128 + (wave & 1) * 64;   // B-col (l-bin) base

    const float* p1 = x1 + (size_t)b * NPIX;
    const float* p2 = x2 + (size_t)b * NPIX;

    const int m = lane & 15;
    float binA[4], binB[4];
#pragma unroll
    for (int t = 0; t < 4; ++t) {
        binA[t] = (float)(k0 + t * 16 + m);         // bins[k] == k exactly
        binB[t] = (float)(l0 + t * 16 + m);
    }

    // issue async loads for one chunk into LDS buffer `buf`
    // (each wave issues exactly 4 global_load_async_to_lds_b32 per chunk)
    auto stage = [&](int cbase, int buf) {
        for (int i = threadIdx.x; i < CHUNK; i += 128) {
            const float* g1 = p1 + cbase + i;
            const float* g2 = p2 + cbase + i;
            unsigned lo1 = (unsigned)(size_t)(__attribute__((address_space(3))) const float*)&lv1[buf][i];
            unsigned lo2 = (unsigned)(size_t)(__attribute__((address_space(3))) const float*)&lv2[buf][i];
            asm volatile("global_load_async_to_lds_b32 %0, %1, off"
                         :: "v"(lo1), "v"(g1) : "memory");
            asm volatile("global_load_async_to_lds_b32 %0, %1, off"
                         :: "v"(lo2), "v"(g2) : "memory");
        }
    };

    v8f acc[4][4] = {};

    const int nbeg = slice * SLICEPIX;
    const int nend = nbeg + SLICEPIX;

    stage(nbeg, 0);                                 // prologue: fill buffer 0
    int cur = 0;

    for (int c = nbeg; c < nend; c += CHUNK) {
        const bool has_next = (c + CHUNK) < nend;
        if (has_next) {
            stage(c + CHUNK, cur ^ 1);              // prefetch behind the WMMAs
            // 8 outstanding max; newest 4 are next-chunk -> current is resident
            asm volatile("s_wait_asynccnt 0x4" ::: "memory");
        } else {
            asm volatile("s_wait_asynccnt 0x0" ::: "memory");
        }
        __syncthreads();                            // all waves' cur data visible

        const float* v1c = lv1[cur];
        const float* v2c = lv2[cur];

        for (int s = 0; s < CHUNK; s += 32) {
            // A 16x32 f16 layout: lanes 0-15 hold K={0..7,16..23},
            //                     lanes 16-31 hold K={8..15,24..31}
            const int off = (lane < 16) ? 0 : 8;

            float pv[16];
            v16h af[4];
            {   // pixel values for the A operand (image 1); x255 folded in FMA
                float4 a0 = *(const float4*)&v1c[s + off + 0];
                float4 a1 = *(const float4*)&v1c[s + off + 4];
                float4 a2 = *(const float4*)&v1c[s + off + 16];
                float4 a3 = *(const float4*)&v1c[s + off + 20];
                pv[0]=a0.x; pv[1]=a0.y; pv[2]=a0.z; pv[3]=a0.w;
                pv[4]=a1.x; pv[5]=a1.y; pv[6]=a1.z; pv[7]=a1.w;
                pv[8]=a2.x; pv[9]=a2.y; pv[10]=a2.z; pv[11]=a2.w;
                pv[12]=a3.x; pv[13]=a3.y; pv[14]=a3.z; pv[15]=a3.w;
            }
#pragma unroll
            for (int t = 0; t < 4; ++t) {
#pragma unroll
                for (int j = 0; j < 16; ++j) {
                    float d = __builtin_fmaf(pv[j], 255.0f, -binA[t]);
                    af[t][j] = (_Float16)__builtin_amdgcn_exp2f(C2 * d * d);
                }
            }

            {   // pixel values for the B operand (image 2)
                float4 b0 = *(const float4*)&v2c[s + off + 0];
                float4 b1 = *(const float4*)&v2c[s + off + 4];
                float4 b2 = *(const float4*)&v2c[s + off + 16];
                float4 b3 = *(const float4*)&v2c[s + off + 20];
                pv[0]=b0.x; pv[1]=b0.y; pv[2]=b0.z; pv[3]=b0.w;
                pv[4]=b1.x; pv[5]=b1.y; pv[6]=b1.z; pv[7]=b1.w;
                pv[8]=b2.x; pv[9]=b2.y; pv[10]=b2.z; pv[11]=b2.w;
                pv[12]=b3.x; pv[13]=b3.y; pv[14]=b3.z; pv[15]=b3.w;
            }
#pragma unroll
            for (int u = 0; u < 4; ++u) {
                v16h bf;
#pragma unroll
                for (int j = 0; j < 16; ++j) {
                    float d = __builtin_fmaf(pv[j], 255.0f, -binB[u]);
                    bf[j] = (_Float16)__builtin_amdgcn_exp2f(C2 * d * d);
                }
#pragma unroll
                for (int t = 0; t < 4; ++t) {
                    acc[t][u] = __builtin_amdgcn_wmma_f32_16x16x32_f16(
                        false, af[t], false, bf, (short)0, acc[t][u], false, false);
                }
            }
        }
        __syncthreads();        // everyone done reading buf[cur] before reuse
        cur ^= 1;
    }

    // C/D layout: VGPR r, lanes 0-15 -> row r, lanes 16-31 -> row r+8; col = lane&15
    const int row0 = (lane < 16) ? 0 : 8;
    const int col  = lane & 15;
    float* J = jpart + slice * JSZ + (size_t)b * NBINS * NBINS;
#pragma unroll
    for (int t = 0; t < 4; ++t) {
#pragma unroll
        for (int u = 0; u < 4; ++u) {
#pragma unroll
            for (int r = 0; r < 8; ++r) {
                J[(size_t)(k0 + t * 16 + row0 + r) * NBINS + (l0 + u * 16 + col)] =
                    acc[t][u][r];
            }
        }
    }
}

// ---------------------------------------------------------------------------
// Kernel 1b: fold slice partials into slice 0 (element-wise, deterministic).
// ---------------------------------------------------------------------------
__global__ __launch_bounds__(256)
void mi_reduce_partials(float* __restrict__ jp)
{
    size_t i = (size_t)blockIdx.x * 256 + threadIdx.x;   // i < JSZ
    float v = jp[i];
#pragma unroll
    for (int s = 1; s < NSLICE; ++s) v += jp[i + s * JSZ];
    jp[i] = v;
}

// ---------------------------------------------------------------------------
// Kernel 2: unnormalized marginal pdfs  pdf[b,k] = mean_n exp2(C2 (v-k)^2)
// ---------------------------------------------------------------------------
__global__ __launch_bounds__(256)
void mi_marginals(const float* __restrict__ x1, const float* __restrict__ x2,
                  float* __restrict__ pdf1, float* __restrict__ pdf2)
{
    __shared__ float r1[256];
    __shared__ float r2[256];
    const int k = blockIdx.x;
    const int b = blockIdx.y;
    const float bin = (float)k;
    const float* p1 = x1 + (size_t)b * NPIX;
    const float* p2 = x2 + (size_t)b * NPIX;

    float s1 = 0.0f, s2 = 0.0f;
    for (int n = threadIdx.x; n < NPIX; n += 256) {
        float d1 = __builtin_fmaf(p1[n], 255.0f, -bin);
        float d2 = __builtin_fmaf(p2[n], 255.0f, -bin);
        s1 += __builtin_amdgcn_exp2f(C2 * d1 * d1);
        s2 += __builtin_amdgcn_exp2f(C2 * d2 * d2);
    }
    r1[threadIdx.x] = s1;
    r2[threadIdx.x] = s2;
    __syncthreads();
    for (int off = 128; off > 0; off >>= 1) {
        if ((int)threadIdx.x < off) {
            r1[threadIdx.x] += r1[threadIdx.x + off];
            r2[threadIdx.x] += r2[threadIdx.x + off];
        }
        __syncthreads();
    }
    if (threadIdx.x == 0) {
        pdf1[b * NBINS + k] = r1[0] * (1.0f / (float)NPIX);
        pdf2[b * NBINS + k] = r2[0] * (1.0f / (float)NPIX);
    }
}

// ---------------------------------------------------------------------------
// Kernel 3: normalize pdfs/joint, entropies, normalized MI.  1 block / batch.
// ---------------------------------------------------------------------------
__global__ __launch_bounds__(256)
void mi_finalize(const float* __restrict__ joint,
                 const float* __restrict__ pdf1, const float* __restrict__ pdf2,
                 float* __restrict__ out)
{
    __shared__ float red[256];
    const int b = blockIdx.x;
    const int t = threadIdx.x;

    auto block_sum = [&](float v) -> float {
        red[t] = v;
        __syncthreads();
        for (int off = 128; off > 0; off >>= 1) {
            if (t < off) red[t] += red[t + off];
            __syncthreads();
        }
        float r = red[0];
        __syncthreads();
        return r;
    };

    float p1 = pdf1[b * NBINS + t];
    float s1 = block_sum(p1);
    float q1 = p1 / (s1 + EPSF);
    float h1 = -block_sum(q1 * __log2f(q1 + EPSF));

    float p2 = pdf2[b * NBINS + t];
    float s2 = block_sum(p2);
    float q2 = p2 / (s2 + EPSF);
    float h2 = -block_sum(q2 * __log2f(q2 + EPSF));

    const float* J = joint + (size_t)b * NBINS * NBINS;
    float sj = 0.0f;
    for (int i = t; i < NBINS * NBINS; i += 256) sj += J[i];
    float S = block_sum(sj);
    float inv = 1.0f / (S + EPSF);
    float hacc = 0.0f;
    for (int i = t; i < NBINS * NBINS; i += 256) {
        float q = J[i] * inv;
        hacc += q * __log2f(q + EPSF);
    }
    float h12 = -block_sum(hacc);

    if (t == 0) {
        float mi = h1 + h2 - h12;
        out[b] = 2.0f * mi / (h1 + h2);   // NORMALIZE = True
    }
}

// ---------------------------------------------------------------------------
extern "C" void kernel_launch(void* const* d_in, const int* in_sizes, int n_in,
                              void* d_out, int out_size, void* d_ws, size_t ws_size,
                              hipStream_t stream)
{
    const float* x1 = (const float*)d_in[0];
    const float* x2 = (const float*)d_in[1];
    // d_in[2] = bins; linspace(0,255,256) == integer k, folded into the kernels.

    float* jp   = (float*)d_ws;                 // NSLICE * 8*256*256 f32 = 8 MB
    float* pdf1 = jp + NSLICE * JSZ;            // 8*256 f32
    float* pdf2 = pdf1 + (size_t)NBATCH * NBINS;
    float* out  = (float*)d_out;                // (8,)

    dim3 gGemm(2, 2, NBATCH * NSLICE);          // l-tiles, k-tiles, batch*slice
    mi_joint_gemm<<<gGemm, 128, 0, stream>>>(x1, x2, jp);

    mi_reduce_partials<<<(int)(JSZ / 256), 256, 0, stream>>>(jp);

    dim3 gMarg(NBINS, NBATCH);
    mi_marginals<<<gMarg, 256, 0, stream>>>(x1, x2, pdf1, pdf2);

    mi_finalize<<<NBATCH, 256, 0, stream>>>(jp, pdf1, pdf2, out);
}
